// TorchElementSpecificN3_72378788872241
// MI455X (gfx1250) — compile-verified, matrix operand order checked by reading, not run on hardware
//
#include <hip/hip_runtime.h>

typedef unsigned short u16;
typedef unsigned int   u32;
typedef __attribute__((ext_vector_type(16))) __bf16 v16bf;
typedef __attribute__((ext_vector_type(8)))  float  v8f;

__device__ __forceinline__ u16 f2bf(float f){
  u32 u = __float_as_uint(f);
  u32 r = (u + 0x7FFFu + ((u >> 16) & 1u)) >> 16;   // round-to-nearest-even
  return (u16)r;
}
__device__ __forceinline__ float bf2f(u16 h){
  return __uint_as_float(((u32)h) << 16);
}
__device__ __forceinline__ float celu1(float x){    // jax.nn.celu, alpha=1
  return x > 0.0f ? x : (__expf(x) - 1.0f);
}

union FragBF { uint4 u[2]; v16bf v; };

// ---------------- fp32 -> bf16 converts ----------------
__global__ void k_convert_x(const float* __restrict__ xr, const float* __restrict__ xt,
                            u16* __restrict__ X, int nr, int total){
  int i = blockIdx.x * blockDim.x + threadIdx.x;
  if (i < total) X[i] = f2bf(i < nr ? xr[i] : xt[i - nr]);
}

// W: [E][K][N] fp32  ->  Wt: [E][N][K] bf16  (NT layout for WMMA B fragments)
__global__ void k_convert_wt(const float* __restrict__ W, u16* __restrict__ Wt,
                             int K, int N, int total){
  int i = blockIdx.x * blockDim.x + threadIdx.x;
  if (i >= total) return;
  int k = i % K; int t = i / K; int n = t % N; int e = t / N;
  Wt[i] = f2bf(W[(size_t)e * K * N + (size_t)k * N + n]);
}

// ---------------- bf16 WMMA GEMM + bias + celu ----------------
// C[M][N] = celu(A[M][K] @ Bt[N][K]^T + bias), bf16 in/out, f32 accumulate.
// Block = 256 threads (8 wave32), block tile 256x128, wave tile 64x64, K-step 32.
#define BM 256
#define BN 128
#define BK 32
#define LDP 40   // padded LDS row stride (elements) to break bank conflicts

__global__ void __launch_bounds__(256)
k_gemm_bf16_celu(const u16* __restrict__ A,     // [M][K] row-major bf16
                 const u16* __restrict__ Bt,    // [N][K] row-major bf16
                 const float* __restrict__ bias,// [N]
                 u16* __restrict__ C,           // [M][N] bf16 out
                 int M, int N, int K)
{
  __shared__ __align__(16) u16 As[BM * LDP];   // 20.0 KB
  __shared__ __align__(16) u16 Bs[BN * LDP];   // 10.0 KB

  const int t    = threadIdx.x;
  const int row0 = blockIdx.y * BM;
  const int col0 = blockIdx.x * BN;

  const int w     = t >> 5;        // wave id (wave32)
  const int lane  = t & 31;
  const int mrow  = lane & 15;     // row (A) / col (B) within 16
  const int khalf = lane >> 4;     // selects K-halves per ISA 16-bit layout
  const int waveM = (w >> 1) * 64; // 4x2 wave grid -> 64x64 per wave
  const int waveN = (w & 1) * 64;

  v8f acc[4][4];
  v8f z = {0.f,0.f,0.f,0.f,0.f,0.f,0.f,0.f};
  #pragma unroll
  for (int tm=0; tm<4; ++tm)
    #pragma unroll
    for (int tn=0; tn<4; ++tn) acc[tm][tn] = z;

  for (int k0 = 0; k0 < K; k0 += BK){
    // cooperative tile load: A = 1024 x 16B chunks (4/thread), B = 512 (2/thread)
    #pragma unroll
    for (int i = 0; i < 4; ++i){
      int lin = t + i * 256;
      int r = lin >> 2, s = lin & 3;
      *(uint4*)&As[r*LDP + s*8] = *(const uint4*)(A + (size_t)(row0 + r)*K + k0 + s*8);
    }
    #pragma unroll
    for (int i = 0; i < 2; ++i){
      int lin = t + i * 256;
      int r = lin >> 2, s = lin & 3;
      *(uint4*)&Bs[r*LDP + s*8] = *(const uint4*)(Bt + (size_t)(col0 + r)*K + k0 + s*8);
    }
    __syncthreads();

    // fragment loads: per lane, K = [8*khalf..+7] and [16+8*khalf..+7] -> 2x b128
    FragBF a[4], b[4];
    #pragma unroll
    for (int tm=0; tm<4; ++tm){
      int r = waveM + tm*16 + mrow;
      a[tm].u[0] = *(const uint4*)&As[r*LDP + 8*khalf];
      a[tm].u[1] = *(const uint4*)&As[r*LDP + 16 + 8*khalf];
    }
    #pragma unroll
    for (int tn=0; tn<4; ++tn){
      int r = waveN + tn*16 + mrow;
      b[tn].u[0] = *(const uint4*)&Bs[r*LDP + 8*khalf];
      b[tn].u[1] = *(const uint4*)&Bs[r*LDP + 16 + 8*khalf];
    }

    // 16 WMMAs per K-step per wave (A frags reused across 4 B frags)
    #pragma unroll
    for (int tm=0; tm<4; ++tm)
      #pragma unroll
      for (int tn=0; tn<4; ++tn)
        acc[tm][tn] = __builtin_amdgcn_wmma_f32_16x16x32_bf16(
            false, a[tm].v, false, b[tn].v, (short)0, acc[tm][tn], false, false);
    __syncthreads();
  }

  // epilogue: C/D layout -> m = base + r + 8*khalf, n = base + (lane&15)
  #pragma unroll
  for (int tm=0; tm<4; ++tm){
    #pragma unroll
    for (int tn=0; tn<4; ++tn){
      int n = col0 + waveN + tn*16 + mrow;
      float bv = bias[n];
      #pragma unroll
      for (int r=0; r<8; ++r){
        int m = row0 + waveM + tm*16 + khalf*8 + r;
        C[(size_t)m * N + n] = f2bf(celu1(acc[tm][tn][r] + bv));
      }
    }
  }
}

// ---------------- layer 3: 1024 -> 1 matvec (linear) ----------------
__global__ void __launch_bounds__(256)
k_matvec3(const u16* __restrict__ H, const float* __restrict__ w3,
          const float* __restrict__ b3, int e, float* __restrict__ out, int K)
{
  int w = threadIdx.x >> 5, lane = threadIdx.x & 31;
  int row = blockIdx.x * 8 + w;
  const u16* hp = H + (size_t)row * K;
  float s = 0.f;
  for (int k = lane; k < K; k += 32) s += bf2f(hp[k]) * w3[k];
  #pragma unroll
  for (int off = 16; off > 0; off >>= 1) s += __shfl_xor(s, off, 32);
  if (lane == 0) out[row] = s + b3[e];
}

// ---------------- exp-kernel regression over reference atoms ----------------
__global__ void __launch_bounds__(256)
k_kernel_regress(const float* __restrict__ Hall, const float* __restrict__ y_ref,
                 const int* __restrict__ l_ref, const int* __restrict__ l_train,
                 const float* __restrict__ alpha, float* __restrict__ out,
                 int Nref, int Ntr, int Mtot)
{
  __shared__ float sn[256], sd[256];
  int i = blockIdx.x, t = threadIdx.x;
  int label = l_train[i];
  int e = label - 1;
  float a = alpha[e];
  const float* He = Hall + (size_t)e * Mtot;
  float ht = He[Nref + i];
  float num = 0.f, den = 0.f;
  for (int j = t; j < Nref; j += 256){
    if (l_ref[j] == label){
      float p = __expf(-a * fabsf(ht - He[j]));   // 1-D embedding: d = |ht-hr|
      num += p * y_ref[j];
      den += p;
    }
  }
  sn[t] = num; sd[t] = den;
  __syncthreads();
  for (int s = 128; s > 0; s >>= 1){
    if (t < s){ sn[t] += sn[t+s]; sd[t] += sd[t+s]; }
    __syncthreads();
  }
  if (t == 0){
    out[i]       = (float)label;   // l_train half of tuple output
    out[Ntr + i] = sn[0] / sd[0];  // y half
  }
}

extern "C" void kernel_launch(void* const* d_in, const int* in_sizes, int n_in,
                              void* d_out, int out_size, void* d_ws, size_t ws_size,
                              hipStream_t stream)
{
  const float* x_ref   = (const float*)d_in[0];
  const float* y_ref   = (const float*)d_in[1];
  const int*   l_ref   = (const int*)  d_in[2];
  const float* x_train = (const float*)d_in[3];
  const int*   l_train = (const int*)  d_in[4];
  const float* W1      = (const float*)d_in[5];
  const float* b1      = (const float*)d_in[6];
  const float* W2      = (const float*)d_in[7];
  const float* b2      = (const float*)d_in[8];
  const float* W3      = (const float*)d_in[9];
  const float* b3      = (const float*)d_in[10];
  const float* alpha   = (const float*)d_in[11];

  const int NREF = 4096, NTR = 8192, MTOT = 12288, D = 384, H = 1024, E = 4;

  // workspace layout (~71.5 MB total)
  char* ws = (char*)d_ws;
  size_t off = 0;
  u16* Xall  = (u16*)(ws + off); off += (size_t)MTOT * D * 2;   // ref||train bf16
  u16* W1t   = (u16*)(ws + off); off += (size_t)E * H * D * 2;  // [E][1024][384]
  u16* W2t   = (u16*)(ws + off); off += (size_t)E * H * H * 2;  // [E][1024][1024]
  u16* H1    = (u16*)(ws + off); off += (size_t)MTOT * H * 2;   // reused per element
  u16* H2    = (u16*)(ws + off); off += (size_t)MTOT * H * 2;
  float* Hall = (float*)(ws + off);                              // [E][12288]

  { int tot = MTOT * D; k_convert_x <<<(tot+255)/256, 256, 0, stream>>>(x_ref, x_train, Xall, NREF*D, tot); }
  { int tot = E*H*D;    k_convert_wt<<<(tot+255)/256, 256, 0, stream>>>(W1, W1t, D, H, tot); }
  { int tot = E*H*H;    k_convert_wt<<<(tot+255)/256, 256, 0, stream>>>(W2, W2t, H, H, tot); }

  dim3 g1(H / BN, MTOT / BM);   // (8, 48)
  for (int e = 0; e < E; ++e){
    k_gemm_bf16_celu<<<g1, 256, 0, stream>>>(Xall, W1t + (size_t)e*H*D, b1 + e*H, H1, MTOT, H, D);
    k_gemm_bf16_celu<<<g1, 256, 0, stream>>>(H1,   W2t + (size_t)e*H*H, b2 + e*H, H2, MTOT, H, H);
    k_matvec3<<<MTOT/8, 256, 0, stream>>>(H2, W3 + e*H, b3, e, Hall + (size_t)e*MTOT, H);
  }

  k_kernel_regress<<<NTR, 256, 0, stream>>>(Hall, y_ref, l_ref, l_train, alpha,
                                            (float*)d_out, NREF, NTR, MTOT);
}
